// AirTransPredictHead_58025008169411
// MI455X (gfx1250) — matrix operand order repaired and checked by reading, not compiled, not run on hardware
//
#include <hip/hip_runtime.h>
#include <hip/hip_bf16.h>
#include <math.h>

// ---------------------------------------------------------------------------
// AirTransPredictHead for MI455X (gfx1250, wave32, WMMA bf16).
// Big GEMMs (query@W1 x2, 1x1 conv) run on v_wmma_f32_16x16x32_bf16 with
// fp32->bf16 conversion fused into the global->LDS stage. All reductions are
// fixed-order (deterministic). Workspace requirement: ~52 MB.
// ---------------------------------------------------------------------------

typedef __bf16 bf16_t;
typedef bf16_t v16bf __attribute__((ext_vector_type(16)));
typedef bf16_t v8bf  __attribute__((ext_vector_type(8)));
typedef float  v8f   __attribute__((ext_vector_type(8)));

#define WAY 5
#define ROI 7
#define CCH 256
#define N1  1024
#define BOX 4096
#define HID 64
#define CHW 12544          // C*ROI*ROI
#define HW  49
#define PP  (BOX*HW)       // 200704 spatial rows for the 1x1 conv GEMM
#define STATS_BLOCKS 512
#define STATS_ROWS   392   // 512*392 == 200704

__device__ __forceinline__ float leaky(float v) { return v > 0.f ? v : 0.01f * v; }

__device__ __forceinline__ unsigned short f2bf(float f) {
  unsigned int u = __float_as_uint(f);
  u += 0x7FFFu + ((u >> 16) & 1u);            // round-to-nearest-even
  return (unsigned short)(u >> 16);
}
__device__ __forceinline__ unsigned int pk2(float a, float b) {
  return (unsigned int)f2bf(a) | ((unsigned int)f2bf(b) << 16);
}
__device__ __forceinline__ float wave_sum(float v) {
  for (int s = 16; s; s >>= 1) v += __shfl_xor(v, s);
  return v;
}

// ---------------------------------------------------------------------------
// WMMA GEMM:  C[M,N] = A[M,K] @ B + bias  (fp32 in/out, bf16 matrix cores)
// MODE 0: A dense row-major [M,K]; B k-major [K,N]        (FC1 GEMMs)
// MODE 1: A gathered: row p=(n*49+hw) -> query[n*CHW + k*HW + hw];
//         B n-major [N,K] (conv_W is [64,256])             (1x1 conv GEMM)
// Block tile 128(M) x 64(N), K-step 32, 8 waves -> each wave 32x32 (4 frags).
// ---------------------------------------------------------------------------
#define A_ST 40   // LDS row stride (bf16 elems), padded vs 32 to dodge banks
#define B_ST 40   // LDS per-column stride for transposed B

template<int MODE>
__global__ __launch_bounds__(256) void gemm_bf16_wmma(
    const float* __restrict__ A, const float* __restrict__ B,
    const float* __restrict__ bias, float* __restrict__ C,
    int K, int N)
{
  __shared__ __align__(16) unsigned short aLDS[128 * A_ST];
  __shared__ __align__(16) unsigned short bLDS[64 * B_ST];

  const int t    = threadIdx.x;
  const int lane = t & 31, wid = t >> 5;
  const int wm   = wid & 3, wn = wid >> 2;       // 4 x 2 wave grid
  const int h    = lane >> 4, ml = lane & 15;
  const int blockM = blockIdx.x * 128;
  const int n0     = blockIdx.y * 64;

  // A global-load mapping: 2 threads per row, 16 consecutive k each
  const int arow = t >> 1;
  const int aks  = (t & 1) << 4;
  const float* aPtr;
  if (MODE == 0) {
    aPtr = A + (size_t)(blockM + arow) * K + aks;
  } else {
    int rowG = blockM + arow;
    int nb = rowG / HW, hw = rowG - nb * HW;
    aPtr = A + (size_t)nb * CHW + hw;            // element c at aPtr[c*HW]
  }

  v8f acc[2][2];
  for (int i = 0; i < 2; ++i)
    for (int j = 0; j < 2; ++j)
      for (int r = 0; r < 8; ++r) acc[i][j][r] = 0.f;

  // Fragment LDS base pointers (layouts per CDNA5 ISA 7.12.2)
  const unsigned short* apL = aLDS + (wm * 32 + ml) * A_ST + h * 8;
  const unsigned short* bpL = bLDS + (wn * 32 + ml) * B_ST + h * 8;

  for (int k0 = 0; k0 < K; k0 += 32) {
    // ---- global loads into registers ----
    float fa[16];
    if (MODE == 0) {
      const float4* p = (const float4*)(aPtr + k0);
      for (int i = 0; i < 4; ++i) {
        float4 q = p[i];
        fa[4*i+0] = q.x; fa[4*i+1] = q.y; fa[4*i+2] = q.z; fa[4*i+3] = q.w;
      }
    } else {
      for (int j = 0; j < 16; ++j) fa[j] = aPtr[(size_t)(k0 + aks + j) * HW];
    }
    float4 fb[2]; int bkk[2], bnn[2];
    for (int c = 0; c < 2; ++c) {
      int chunk = t * 2 + c;                     // 512 float4 chunks of B tile
      if (MODE == 0) {
        int kk = chunk >> 4, nq = chunk & 15;
        bkk[c] = kk; bnn[c] = nq * 4;
        fb[c] = *(const float4*)(B + (size_t)(k0 + kk) * N + n0 + nq * 4);
      } else {
        int nn = chunk >> 3, kq = chunk & 7;
        bkk[c] = kq * 4; bnn[c] = nn;
        fb[c] = *(const float4*)(B + (size_t)nn * K + k0 + kq * 4);
      }
    }
    __syncthreads();
    // ---- convert + store to LDS ----
    {
      uint4 u0, u1;
      u0.x = pk2(fa[0], fa[1]);  u0.y = pk2(fa[2], fa[3]);
      u0.z = pk2(fa[4], fa[5]);  u0.w = pk2(fa[6], fa[7]);
      u1.x = pk2(fa[8], fa[9]);  u1.y = pk2(fa[10], fa[11]);
      u1.z = pk2(fa[12], fa[13]); u1.w = pk2(fa[14], fa[15]);
      uint4* d = (uint4*)(aLDS + arow * A_ST + aks);
      d[0] = u0; d[1] = u1;
    }
    for (int c = 0; c < 2; ++c) {
      if (MODE == 0) {          // scatter 4 bf16 down a transposed column
        bLDS[(bnn[c] + 0) * B_ST + bkk[c]] = f2bf(fb[c].x);
        bLDS[(bnn[c] + 1) * B_ST + bkk[c]] = f2bf(fb[c].y);
        bLDS[(bnn[c] + 2) * B_ST + bkk[c]] = f2bf(fb[c].z);
        bLDS[(bnn[c] + 3) * B_ST + bkk[c]] = f2bf(fb[c].w);
      } else {                  // conv_W already [n][k]: contiguous store
        uint2 u; u.x = pk2(fb[c].x, fb[c].y); u.y = pk2(fb[c].z, fb[c].w);
        *(uint2*)(bLDS + bnn[c] * B_ST + bkk[c]) = u;
      }
    }
    __syncthreads();
    // ---- fragments + WMMA ----
    v8bf al0 = *(const v8bf*)(apL);
    v8bf ah0 = *(const v8bf*)(apL + 16);
    v8bf al1 = *(const v8bf*)(apL + 16 * A_ST);
    v8bf ah1 = *(const v8bf*)(apL + 16 * A_ST + 16);
    v16bf afr[2];
    afr[0] = __builtin_shufflevector(al0, ah0, 0,1,2,3,4,5,6,7,8,9,10,11,12,13,14,15);
    afr[1] = __builtin_shufflevector(al1, ah1, 0,1,2,3,4,5,6,7,8,9,10,11,12,13,14,15);
    v8bf bl0 = *(const v8bf*)(bpL);
    v8bf bh0 = *(const v8bf*)(bpL + 16);
    v8bf bl1 = *(const v8bf*)(bpL + 16 * B_ST);
    v8bf bh1 = *(const v8bf*)(bpL + 16 * B_ST + 16);
    v16bf bfr[2];
    bfr[0] = __builtin_shufflevector(bl0, bh0, 0,1,2,3,4,5,6,7,8,9,10,11,12,13,14,15);
    bfr[1] = __builtin_shufflevector(bl1, bh1, 0,1,2,3,4,5,6,7,8,9,10,11,12,13,14,15);
    for (int i = 0; i < 2; ++i)
      for (int j = 0; j < 2; ++j)
        acc[i][j] = __builtin_amdgcn_wmma_f32_16x16x32_bf16(
            false, afr[i], false, bfr[j], (short)0, acc[i][j], false, false);
  }

  // ---- epilogue: + bias, fp32 store ----
  for (int j = 0; j < 2; ++j) {
    int col = n0 + wn * 32 + j * 16 + ml;
    float bv = bias[col];
    for (int i = 0; i < 2; ++i) {
      int rbase = blockM + wm * 32 + i * 16 + h * 8;
      for (int r = 0; r < 8; ++r)
        C[(size_t)(rbase + r) * N + col] = acc[i][j][r] + bv;
    }
  }
}

// ---------------------------------------------------------------------------
// FC2 + row-mean: qmean[n] = mean_j leaky(X1[n,:] @ W2[:,j] + b2[j])
// ---------------------------------------------------------------------------
template<int NOUT>
__global__ __launch_bounds__(256) void qmean_kernel(
    const float* __restrict__ X1, const float* __restrict__ W2,
    const float* __restrict__ b2, float* __restrict__ outMean)
{
  __shared__ float row[N1];
  __shared__ float wred[8];
  int t = threadIdx.x, lane = t & 31, wid = t >> 5;
  const float* src = X1 + (size_t)blockIdx.x * N1;
  for (int k = t; k < N1; k += 256) row[k] = src[k];
  __syncthreads();
  float wacc = 0.f;
  for (int j = wid; j < NOUT; j += 8) {
    float p = 0.f;
    for (int k = lane; k < N1; k += 32) p += row[k] * W2[k * NOUT + j];
    p = wave_sum(p);
    if (lane == 0) wacc += leaky(p + b2[j]);
  }
  if (lane == 0) wred[wid] = wacc;
  __syncthreads();
  if (t == 0) {
    float s = 0.f;
    for (int i = 0; i < 8; ++i) s += wred[i];
    outMean[blockIdx.x] = s / (float)NOUT;
  }
}

// ---------------------------------------------------------------------------
// Support FC1 (M=5, too small for WMMA tiles): one wave per output
// ---------------------------------------------------------------------------
__global__ __launch_bounds__(256) void supfc1_kernel(
    const float* __restrict__ S, const float* __restrict__ W1,
    const float* __restrict__ b1, float* __restrict__ out)   // out [5,1024]
{
  int wid = threadIdx.x >> 5, lane = threadIdx.x & 31;
  int g = blockIdx.x * 8 + wid;            // 0..5119
  int w = g >> 10, n = g & 1023;
  const float* srow = S + (size_t)w * CHW;
  float p = 0.f;
  for (int k = lane; k < CHW; k += 32) p += srow[k] * W1[(size_t)k * N1 + n];
  p = wave_sum(p);
  if (lane == 0) out[g] = p + b1[n];
}

// ---------------------------------------------------------------------------
// Support heads: sf=[5,64], ss=[5,5], means, auxrank(sf) -> loss_f
// ---------------------------------------------------------------------------
__global__ __launch_bounds__(256) void small_heads_kernel(
    const float* __restrict__ sf1, const float* __restrict__ ss1,
    const float* __restrict__ fW2, const float* __restrict__ fb2,
    const float* __restrict__ sW2, const float* __restrict__ sb2,
    float* __restrict__ sf_mean, float* __restrict__ ss_mean,
    float* __restrict__ loss_f)
{
  __shared__ float sf[WAY * 64];
  __shared__ float ss[WAY * 5];
  __shared__ float nrm[WAY];
  __shared__ float lred[64];
  int t = threadIdx.x;
  for (int g = t; g < WAY * 64; g += 256) {
    int w = g >> 6, j = g & 63;
    float a = fb2[j];
    const float* r = sf1 + w * N1;
    for (int k = 0; k < N1; ++k) a += r[k] * fW2[k * 64 + j];
    sf[g] = leaky(a);
  }
  for (int g = t; g < WAY * 5; g += 256) {
    int w = g / 5, j = g % 5;
    float a = sb2[j];
    const float* r = ss1 + w * N1;
    for (int k = 0; k < N1; ++k) a += r[k] * sW2[k * 5 + j];
    ss[g] = leaky(a);
  }
  __syncthreads();
  if (t < WAY) {
    float sm = 0.f, n2 = 0.f;
    for (int j = 0; j < 64; ++j) { float v = sf[t * 64 + j]; sm += v; n2 += v * v; }
    sf_mean[t] = sm / 64.f;
    nrm[t] = fmaxf(sqrtf(n2), 1e-12f);
    float s2 = 0.f;
    for (int j = 0; j < 5; ++j) s2 += ss[t * 5 + j];
    ss_mean[t] = s2 / 5.f;
  }
  __syncthreads();
  if (t < 64) {
    float tot = 0.f, ssq = 0.f;
    for (int w = 0; w < WAY; ++w) {
      float q = sf[w * 64 + t] / nrm[w]; q *= q;
      tot += q; ssq += q * q;
    }
    lred[t] = (tot * tot - ssq) * 0.5f;
  }
  __syncthreads();
  if (t == 0) {
    float s = 0.f;
    for (int j = 0; j < 64; ++j) s += lred[j];
    *loss_f = s / 64.f;
  }
}

// ---------------------------------------------------------------------------
// BatchNorm stats on y [PP,64]: deterministic two-stage reduction
// ---------------------------------------------------------------------------
__global__ __launch_bounds__(256) void bn_partial_kernel(
    const float* __restrict__ Y, float* __restrict__ part)
{
  __shared__ float ls[256], ls2[256];
  int t = threadIdx.x;
  int o = t & 63, pr = t >> 6;
  int p0 = blockIdx.x * STATS_ROWS;
  float s = 0.f, s2 = 0.f;
  for (int p = p0 + pr; p < p0 + STATS_ROWS; p += 4) {
    float v = Y[(size_t)p * 64 + o];
    s += v; s2 += v * v;
  }
  ls[t] = s; ls2[t] = s2;
  __syncthreads();
  if (t < 64) {
    part[blockIdx.x * 128 + t]      = ls[t] + ls[t + 64] + ls[t + 128] + ls[t + 192];
    part[blockIdx.x * 128 + 64 + t] = ls2[t] + ls2[t + 64] + ls2[t + 128] + ls2[t + 192];
  }
}

__global__ __launch_bounds__(64) void bn_final_kernel(
    const float* __restrict__ part, const float* __restrict__ gamma,
    const float* __restrict__ beta, float* __restrict__ coefA,
    float* __restrict__ coefB)
{
  int t = threadIdx.x;                       // 64 threads, one per channel
  float S = 0.f, S2 = 0.f;
  for (int b = 0; b < STATS_BLOCKS; ++b) {
    S  += part[b * 128 + t];
    S2 += part[b * 128 + 64 + t];
  }
  float inv = 1.f / (float)PP;
  float mu  = S * inv;
  float var = S2 * inv - mu * mu;
  float a = gamma[t] * rsqrtf(var + 1e-5f);
  coefA[t] = a;
  coefB[t] = beta[t] - mu * a;
}

__global__ __launch_bounds__(256) void qcmean_kernel(
    const float* __restrict__ Y, const float* __restrict__ coefA,
    const float* __restrict__ coefB, float* __restrict__ outMean)
{
  __shared__ float red[256];
  int t = threadIdx.x;
  size_t base = (size_t)blockIdx.x * HW * 64;
  float acc = 0.f;
  for (int i = t; i < HW * 64; i += 256) {
    int o = i & 63;
    acc += leaky(coefA[o] * Y[base + i] + coefB[o]);
  }
  red[t] = acc; __syncthreads();
  for (int s = 128; s; s >>= 1) { if (t < s) red[t] += red[t + s]; __syncthreads(); }
  if (t == 0) outMean[blockIdx.x] = red[0] / (float)(HW * 64);
}

// ---------------------------------------------------------------------------
// Support conv branch (tiny): raw conv, own BN stats, sc_mean, auxrank -> loss_c
// ---------------------------------------------------------------------------
__global__ __launch_bounds__(256) void ysup_kernel(
    const float* __restrict__ S, const float* __restrict__ Wc,
    const float* __restrict__ bc, float* __restrict__ ysup)   // [5][64][49]
{
  int g = blockIdx.x * 256 + threadIdx.x;
  if (g >= WAY * 64 * HW) return;
  int w = g / (64 * HW);
  int rem = g - w * (64 * HW);
  int o = rem / HW, hw = rem - o * HW;
  float acc = bc[o];
  const float* sp = S + (size_t)w * CHW + hw;
  const float* wp = Wc + o * CCH;
  for (int c = 0; c < CCH; ++c) acc += sp[c * HW] * wp[c];
  ysup[g] = acc;
}

__global__ __launch_bounds__(256) void supfinish_kernel(
    const float* __restrict__ ysup, const float* __restrict__ gamma,
    const float* __restrict__ beta, float* __restrict__ sc_mean,
    float* __restrict__ loss_c)
{
  __shared__ float cA[64], cB[64];
  __shared__ float rn[WAY * HW];
  __shared__ float red[256];
  int t = threadIdx.x;
  if (t < 64) {
    float s = 0.f, s2 = 0.f;
    for (int w = 0; w < WAY; ++w)
      for (int hw = 0; hw < HW; ++hw) {
        float v = ysup[w * 64 * HW + t * HW + hw];
        s += v; s2 += v * v;
      }
    float inv = 1.f / (float)(WAY * HW);
    float mu = s * inv, var = s2 * inv - mu * mu;
    float a = gamma[t] * rsqrtf(var + 1e-5f);
    cA[t] = a; cB[t] = beta[t] - mu * a;
  }
  __syncthreads();
  if (t < WAY) {
    float sm = 0.f;
    for (int o = 0; o < 64; ++o)
      for (int hw = 0; hw < HW; ++hw)
        sm += leaky(cA[o] * ysup[t * 64 * HW + o * HW + hw] + cB[o]);
    sc_mean[t] = sm / (float)(64 * HW);
  }
  if (t < WAY * HW) {       // per-(way,position) channel norm (F.normalize dim=1)
    int w = t / HW, hw = t - w * HW;
    float n2 = 0.f;
    for (int c = 0; c < 64; ++c) {
      float v = leaky(cA[c] * ysup[w * 64 * HW + c * HW + hw] + cB[c]);
      n2 += v * v;
    }
    rn[t] = 1.f / fmaxf(sqrtf(n2), 1e-12f);
  }
  __syncthreads();
  float acc = 0.f;
  for (int j = t; j < 64 * HW; j += 256) {
    int o = j / HW, hw = j - o * HW;
    float tot = 0.f, ssq = 0.f;
    for (int w = 0; w < WAY; ++w) {
      float v = leaky(cA[o] * ysup[w * 64 * HW + o * HW + hw] + cB[o]) * rn[w * HW + hw];
      float q = v * v;
      tot += q; ssq += q * q;
    }
    acc += (tot * tot - ssq) * 0.5f;
  }
  red[t] = acc; __syncthreads();
  for (int s = 128; s; s >>= 1) { if (t < s) red[t] += red[t + s]; __syncthreads(); }
  if (t == 0) *loss_c = red[0] / (float)(64 * HW);
}

// ---------------------------------------------------------------------------
// bbox_deltas = x @ bbox_W + b
// ---------------------------------------------------------------------------
__global__ __launch_bounds__(256) void bbox_kernel(
    const float* __restrict__ x, const float* __restrict__ W,
    const float* __restrict__ b, float* __restrict__ out)
{
  int g = blockIdx.x * 256 + threadIdx.x;    // exactly 98304
  int row = g / 24, j = g - row * 24;
  float acc = b[j];
  const float* xr = x + (size_t)row * N1;
  for (int k = 0; k < N1; ++k) acc += xr[k] * W[k * 24 + j];
  out[g] = acc;
}

// ---------------------------------------------------------------------------
// Final metric head: bg dot, weighted distances, dec layer, scores
// ---------------------------------------------------------------------------
__global__ __launch_bounds__(256) void combine_kernel(
    const float* __restrict__ x, const float* __restrict__ bgW, const float* __restrict__ bgb,
    const float* __restrict__ sfm, const float* __restrict__ scm, const float* __restrict__ ssm,
    const float* __restrict__ qfm, const float* __restrict__ qcm, const float* __restrict__ qsm,
    const float* __restrict__ decW, const float* __restrict__ decb,
    const float* __restrict__ scale, const float* __restrict__ dwv,
    float* __restrict__ out)
{
  int lane = threadIdx.x & 31, wid = threadIdx.x >> 5;
  int box = blockIdx.x * 8 + wid;
  const float* xr = x + (size_t)box * N1;
  float p = 0.f;
  for (int k = lane; k < N1; k += 32) p += xr[k] * bgW[k];
  p = wave_sum(p);
  float bg = p + bgb[0];
  // wts = softmax(exp(distance_weight))
  float u0 = expf(dwv[0]), u1 = expf(dwv[1]), u2 = expf(dwv[2]);
  float m = fmaxf(u0, fmaxf(u1, u2));
  float e0 = expf(u0 - m), e1 = expf(u1 - m), e2 = expf(u2 - m);
  float inv = 1.f / (e0 + e1 + e2);
  float w0 = e0 * inv, w1 = e1 * inv, w2 = e2 * inv;
  float qf = qfm[box], qc = qcm[box], qs = qsm[box];
  float dist[6];
  for (int j = 0; j < 5; ++j) {
    float a = sfm[j] - qf, b = scm[j] - qc, c = ssm[j] - qs;
    dist[j] = w0 * a * a + w1 * b * b + w2 * c * c;
  }
  dist[5] = (w0 + w1 + w2) * bg;
  if (lane < 6) {
    float tj = decb[lane];
    for (int k = 0; k < 6; ++k) tj += dist[k] * decW[k * 6 + lane];
    float l = leaky(tj);
    float d = 1.f / (1.f + expf(-l));
    out[box * 6 + lane] = -(2.f * (d - 0.5f)) * expf(scale[0]);
  }
}

__global__ void loss_kernel(const float* lf, const float* lc, float* out)
{
  if (threadIdx.x == 0 && blockIdx.x == 0) out[0] = lf[0] + lc[0];
}

// ---------------------------------------------------------------------------
extern "C" void kernel_launch(void* const* d_in, const int* in_sizes, int n_in,
                              void* d_out, int out_size, void* d_ws, size_t ws_size,
                              hipStream_t stream)
{
  (void)in_sizes; (void)n_in; (void)out_size; (void)ws_size;
  const float* support  = (const float*)d_in[0];
  const float* query    = (const float*)d_in[1];
  const float* x        = (const float*)d_in[2];
  const float* bbox_W   = (const float*)d_in[3];
  const float* bbox_b   = (const float*)d_in[4];
  const float* conv_W   = (const float*)d_in[5];
  const float* conv_b   = (const float*)d_in[6];
  const float* bn_g     = (const float*)d_in[7];
  const float* bn_b     = (const float*)d_in[8];
  const float* flat_W1  = (const float*)d_in[9];
  const float* flat_b1  = (const float*)d_in[10];
  const float* flat_W2  = (const float*)d_in[11];
  const float* flat_b2  = (const float*)d_in[12];
  const float* score_W1 = (const float*)d_in[13];
  const float* score_b1 = (const float*)d_in[14];
  const float* score_W2 = (const float*)d_in[15];
  const float* score_b2 = (const float*)d_in[16];
  const float* bg_W     = (const float*)d_in[17];
  const float* bg_b     = (const float*)d_in[18];
  const float* dec_W    = (const float*)d_in[19];
  const float* dec_b    = (const float*)d_in[20];
  const float* scale    = (const float*)d_in[21];
  const float* dwv      = (const float*)d_in[22];

  // workspace layout (floats); bigA reused: qf1 -> qs1 -> conv y [PP,64]
  float* ws   = (float*)d_ws;
  float* bigA = ws;
  const size_t SB = (size_t)PP * 64;        // 12,845,056 floats (51.4 MB)
  float* sf1   = ws + SB;
  float* ss1   = sf1 + WAY * N1;
  float* sfm   = ss1 + WAY * N1;
  float* ssm   = sfm + WAY;
  float* scm   = ssm + WAY;
  float* lossf = scm + WAY;
  float* lossc = lossf + 1;
  float* coefA = lossc + 1;
  float* coefB = coefA + 64;
  float* qfm   = coefB + 64;
  float* qsm   = qfm + BOX;
  float* qcm   = qsm + BOX;
  float* ysup  = qcm + BOX;
  float* part  = ysup + WAY * 64 * HW;      // [512][128]

  float* out_scores = (float*)d_out;        // [4096,6]
  float* out_bbox   = out_scores + BOX * 6; // [4096,24]
  float* out_loss   = out_bbox + BOX * 24;  // scalar

  dim3 blk(256);

  // flatten branch: qf1 = query@flat_W1+b1 (WMMA), then FC2+mean
  gemm_bf16_wmma<0><<<dim3(BOX / 128, N1 / 64), blk, 0, stream>>>(
      query, flat_W1, flat_b1, bigA, CHW, N1);
  qmean_kernel<64><<<BOX, blk, 0, stream>>>(bigA, flat_W2, flat_b2, qfm);

  // score branch (reuses bigA)
  gemm_bf16_wmma<0><<<dim3(BOX / 128, N1 / 64), blk, 0, stream>>>(
      query, score_W1, score_b1, bigA, CHW, N1);
  qmean_kernel<5><<<BOX, blk, 0, stream>>>(bigA, score_W2, score_b2, qsm);

  // conv branch: y[200704,64] via gathered-A WMMA GEMM, BN stats, per-box mean
  gemm_bf16_wmma<1><<<dim3(PP / 128, 1), blk, 0, stream>>>(
      query, conv_W, conv_b, bigA, CCH, 64);
  bn_partial_kernel<<<STATS_BLOCKS, blk, 0, stream>>>(bigA, part);
  bn_final_kernel<<<1, 64, 0, stream>>>(part, bn_g, bn_b, coefA, coefB);
  qcmean_kernel<<<BOX, blk, 0, stream>>>(bigA, coefA, coefB, qcm);

  // support paths (tiny)
  supfc1_kernel<<<640, blk, 0, stream>>>(support, flat_W1, flat_b1, sf1);
  supfc1_kernel<<<640, blk, 0, stream>>>(support, score_W1, score_b1, ss1);
  small_heads_kernel<<<1, blk, 0, stream>>>(sf1, ss1, flat_W2, flat_b2,
                                            score_W2, score_b2, sfm, ssm, lossf);
  ysup_kernel<<<62, blk, 0, stream>>>(support, conv_W, conv_b, ysup);
  supfinish_kernel<<<1, blk, 0, stream>>>(ysup, bn_g, bn_b, scm, lossc);

  // outputs
  bbox_kernel<<<384, blk, 0, stream>>>(x, bbox_W, bbox_b, out_bbox);
  combine_kernel<<<512, blk, 0, stream>>>(x, bg_W, bg_b, sfm, scm, ssm,
                                          qfm, qcm, qsm, dec_W, dec_b,
                                          scale, dwv, out_scores);
  loss_kernel<<<1, 1, 0, stream>>>(lossf, lossc, out_loss);
}
